// DiscoveryNet_28776280883565
// MI455X (gfx1250) — matrix operand order repaired and checked by reading, not compiled, not run on hardware
//
#include <hip/hip_runtime.h>
#include <hip/hip_bf16.h>
#include <math.h>

typedef __attribute__((ext_vector_type(16))) _Float16 v16h;
typedef __attribute__((ext_vector_type(8)))  float    v8f;

#define N_NODES        262144
#define N_EDGES        4194304
#define NUM_GRAPHS     1024
#define EDGE_TILES     (N_EDGES / 16)      // 262144 tiles of 16 edges
#define EDGE_BLOCKS    2048
#define WAVES_PER_BLK  8
#define TOTAL_WAVES    (EDGE_BLOCKS * WAVES_PER_BLK)   // 16384 -> 16 tiles/wave exactly

// ---------------------------------------------------------------------------
// workspace zeroing (agg + cnt + pooled are accumulated with atomics)
// ---------------------------------------------------------------------------
__global__ void zero_f32(float* __restrict__ p, int n) {
    int stride = gridDim.x * blockDim.x;
    for (int i = blockIdx.x * blockDim.x + threadIdx.x; i < n; i += stride)
        p[i] = 0.0f;
}

// ---------------------------------------------------------------------------
// Edge MLP on WMMA: 16 edges/tile -> A(16x32 f16) x W(32x16 f16) twice.
// One wave per tile, private LDS slice per wave (wavefront-scope fences only).
// ---------------------------------------------------------------------------
__global__ __launch_bounds__(256) void edge_mlp_wmma(
    const float* __restrict__ x,        // [N_NODES,4]
    const float* __restrict__ pos,      // [N_NODES,3]
    const int*   __restrict__ ei,       // [2,N_EDGES]
    const float* __restrict__ w1,       // [9,16]
    const float* __restrict__ b1,       // [16]
    const float* __restrict__ w2,       // [16,16]
    const float* __restrict__ b2,       // [16]
    float* __restrict__ agg,            // [N_NODES,16] (atomic)
    float* __restrict__ cnt)            // [N_NODES]    (atomic)
{
    __shared__ _Float16 featA[WAVES_PER_BLK][16][32];  // 8 KB : edge features, K-padded
    __shared__ _Float16 hidA [WAVES_PER_BLK][16][16];  // 4 KB : silu(hidden), C->A relayout
    __shared__ int      dstI [WAVES_PER_BLK][16];

    const int lane   = threadIdx.x & 31;
    const int wave   = threadIdx.x >> 5;
    const int gw     = blockIdx.x * WAVES_PER_BLK + wave;
    const int col    = lane & 15;               // N (feature column) / M row for A
    const bool hi    = lane >= 16;
    const int kb     = hi ? 8 : 0;

    // ---- constant B fragments (zero-padded K): lanes 0-15 = K 0..15, lanes 16-31 = K 16..31
    v16h b1f, b2f;
#pragma unroll
    for (int i = 0; i < 16; ++i) {
        float v1 = 0.0f, v2 = 0.0f;
        if (!hi) {
            if (i < 9) v1 = w1[i * 16 + col];
            v2 = w2[i * 16 + col];
        }
        b1f[i] = (_Float16)v1;
        b2f[i] = (_Float16)v2;
    }
    const float bias1 = b1[col];
    const float bias2 = b2[col];

    // ---- K-pad of the feature tile is invariant: zero it once (lanes 0-15 own a row)
    if (!hi) {
#pragma unroll
        for (int k = 9; k < 32; ++k) featA[wave][lane][k] = (_Float16)0.0f;
    }
    __builtin_amdgcn_fence(__ATOMIC_ACQ_REL, "wavefront");

    for (int t = gw; t < EDGE_TILES; t += TOTAL_WAVES) {
        // ---- phase 1: lanes 0-15 each gather one edge into the LDS A-tile
        if (!hi) {
            const int e = t * 16 + lane;
            // cover the grid-stride for the only HBM-streamed input
            if (t + TOTAL_WAVES < EDGE_TILES) {
                __builtin_prefetch(&ei[e + TOTAL_WAVES * 16], 0, 3);
                __builtin_prefetch(&ei[N_EDGES + e + TOTAL_WAVES * 16], 0, 3);
            }
            const int s = ei[e];
            const int d = ei[N_EDGES + e];
            const float dx = pos[3 * s + 0] - pos[3 * d + 0];
            const float dy = pos[3 * s + 1] - pos[3 * d + 1];
            const float dz = pos[3 * s + 2] - pos[3 * d + 2];
            const float dist = __builtin_amdgcn_sqrtf(dx * dx + dy * dy + dz * dz);
            _Float16* row = &featA[wave][lane][0];
            row[0] = (_Float16)x[4 * d + 0];
            row[1] = (_Float16)x[4 * d + 1];
            row[2] = (_Float16)x[4 * d + 2];
            row[3] = (_Float16)x[4 * d + 3];
            row[4] = (_Float16)x[4 * s + 0];
            row[5] = (_Float16)x[4 * s + 1];
            row[6] = (_Float16)x[4 * s + 2];
            row[7] = (_Float16)x[4 * s + 3];
            row[8] = (_Float16)dist;
            dstI[wave][lane] = d;
            atomicAdd(&cnt[d], 1.0f);
        }
        __builtin_amdgcn_fence(__ATOMIC_ACQ_REL, "wavefront");

        // ---- A fragment, layer 1 (lane holds row col; halves K={2kb..2kb+15})
        v16h a1;
#pragma unroll
        for (int i = 0; i < 8; ++i) {
            a1[i]     = featA[wave][col][2 * kb + i];
            a1[8 + i] = featA[wave][col][2 * kb + 8 + i];
        }
        v8f c1 = {0.f, 0.f, 0.f, 0.f, 0.f, 0.f, 0.f, 0.f};
        c1 = __builtin_amdgcn_wmma_f32_16x16x32_f16(false, a1, false, b1f,
                                                    (short)0, c1, false, false);

        // ---- bias + SiLU in f32 (hw exp/rcp), relayout C(M,N) -> LDS for next A
#pragma unroll
        for (int r = 0; r < 8; ++r) {
            float v = c1[r] + bias1;
            v = v * __builtin_amdgcn_rcpf(1.0f + __expf(-v));   // silu
            hidA[wave][r + kb][col] = (_Float16)v;              // M = r+kb, N = col
        }
        __builtin_amdgcn_fence(__ATOMIC_ACQ_REL, "wavefront");

        // ---- A fragment, layer 2 (K = 16 real + 16 zero pad)
        v16h a2;
#pragma unroll
        for (int i = 0; i < 8; ++i) {
            a2[i]     = hi ? (_Float16)0.0f : hidA[wave][col][i];
            a2[8 + i] = hi ? (_Float16)0.0f : hidA[wave][col][8 + i];
        }
        v8f c2 = {0.f, 0.f, 0.f, 0.f, 0.f, 0.f, 0.f, 0.f};
        c2 = __builtin_amdgcn_wmma_f32_16x16x32_f16(false, a2, false, b2f,
                                                    (short)0, c2, false, false);

        // ---- scatter: lane=feature col, VGPR r = edge row -> 8 atomics/lane
#pragma unroll
        for (int r = 0; r < 8; ++r) {
            const int m  = r + kb;
            const int dn = dstI[wave][m];
            atomicAdd(&agg[dn * 16 + col], c2[r] + bias2);
        }
        __builtin_amdgcn_fence(__ATOMIC_ACQ_REL, "wavefront"); // before featA rewrite
    }
}

// ---------------------------------------------------------------------------
// Node stage: h = relu(agg/max(cnt,1)), s = softmax(h @ pool_w + pool_b),
// pooled[batch] += s_k * h  (atomics into 128 KB L2-resident buffer)
// ---------------------------------------------------------------------------
__global__ __launch_bounds__(256) void node_pool(
    const float* __restrict__ agg, const float* __restrict__ cnt,
    const int*   __restrict__ batch,
    const float* __restrict__ pw, const float* __restrict__ pb,
    float* __restrict__ s_out,        // [N_NODES,2]
    float* __restrict__ pooled)       // [NUM_GRAPHS,2,16] (atomic)
{
    const int n = blockIdx.x * blockDim.x + threadIdx.x;
    if (n >= N_NODES) return;
    const float inv = __builtin_amdgcn_rcpf(fmaxf(cnt[n], 1.0f));
    float h[16];
    float l0 = pb[0], l1 = pb[1];
#pragma unroll
    for (int i = 0; i < 16; ++i) {
        h[i] = fmaxf(agg[n * 16 + i] * inv, 0.0f);
        l0 += h[i] * pw[i * 2 + 0];
        l1 += h[i] * pw[i * 2 + 1];
    }
    const float m  = fmaxf(l0, l1);
    const float e0 = __expf(l0 - m), e1 = __expf(l1 - m);
    const float rs = __builtin_amdgcn_rcpf(e0 + e1);
    const float s0 = e0 * rs, s1 = e1 * rs;
    s_out[2 * n + 0] = s0;
    s_out[2 * n + 1] = s1;
    float* pg = pooled + batch[n] * 32;
#pragma unroll
    for (int i = 0; i < 16; ++i) {
        atomicAdd(&pg[i],      s0 * h[i]);
        atomicAdd(&pg[16 + i], s1 * h[i]);
    }
}

// ---------------------------------------------------------------------------
// Graph head: z, recon, pair potential V and ANALYTIC forces = -dV/dq.
// V = sum_j softplus(d*w1_j + b1_j)*w2_j + b2 ; dV/dd = sum_j sig(.)*w1_j*w2_j
// d = sqrt(|q0-q1|^2 + 1e-6); dd/dq0 = diff/d.
// ---------------------------------------------------------------------------
__global__ __launch_bounds__(256) void graph_head(
    const float* __restrict__ pooled,
    const float* __restrict__ tw, const float* __restrict__ tb,     // toz 16x4
    const float* __restrict__ vw1, const float* __restrict__ vb1,   // 1x32
    const float* __restrict__ vw2, const float* __restrict__ vb2,   // 32x1
    const float* __restrict__ bw, const float* __restrict__ bb,     // 8x32
    float* __restrict__ recon_out,   // [1024,8,4]
    float* __restrict__ z_out,       // [1024,2,4]
    float* __restrict__ forces_out,  // [1024,2,2]
    float* __restrict__ V_out)       // [1024,1]
{
    const int g = blockIdx.x * blockDim.x + threadIdx.x;
    if (g >= NUM_GRAPHS) return;

    float zf[8];
#pragma unroll
    for (int k = 0; k < 2; ++k)
#pragma unroll
        for (int d = 0; d < 4; ++d) {
            float acc = tb[d];
#pragma unroll
            for (int i = 0; i < 16; ++i)
                acc += pooled[g * 32 + k * 16 + i] * tw[i * 4 + d];
            zf[k * 4 + d] = acc;
        }
#pragma unroll
    for (int j = 0; j < 8; ++j) z_out[g * 8 + j] = zf[j];

#pragma unroll
    for (int j = 0; j < 32; ++j) {
        float a = bb[j];
#pragma unroll
        for (int c = 0; c < 8; ++c) a += zf[c] * bw[c * 32 + j];
        recon_out[g * 32 + j] = a;
    }

    const float dx = zf[0] - zf[4];
    const float dy = zf[1] - zf[5];
    const float d  = __builtin_amdgcn_sqrtf(dx * dx + dy * dy + 1e-6f);
    float V = vb2[0], dVdd = 0.0f;
#pragma unroll
    for (int j = 0; j < 32; ++j) {
        const float t  = d * vw1[j] + vb1[j];
        const float sp = (t > 20.0f) ? t : __logf(1.0f + __expf(t));
        const float sg = __builtin_amdgcn_rcpf(1.0f + __expf(-t));
        V    += sp * vw2[j];
        dVdd += sg * vw1[j] * vw2[j];
    }
    const float invd = __builtin_amdgcn_rcpf(d);
    const float fx = dVdd * dx * invd;
    const float fy = dVdd * dy * invd;
    forces_out[g * 4 + 0] = -fx;
    forces_out[g * 4 + 1] = -fy;
    forces_out[g * 4 + 2] =  fx;
    forces_out[g * 4 + 3] =  fy;
    V_out[g] = V;
}

// ---------------------------------------------------------------------------
extern "C" void kernel_launch(void* const* d_in, const int* in_sizes, int n_in,
                              void* d_out, int out_size, void* d_ws, size_t ws_size,
                              hipStream_t stream) {
    const float* x      = (const float*)d_in[0];
    const float* pos    = (const float*)d_in[1];
    const int*   ei     = (const int*)  d_in[2];
    const int*   batch  = (const int*)  d_in[3];
    const float* enc_w1 = (const float*)d_in[4];
    const float* enc_b1 = (const float*)d_in[5];
    const float* enc_w2 = (const float*)d_in[6];
    const float* enc_b2 = (const float*)d_in[7];
    const float* pool_w = (const float*)d_in[8];
    const float* pool_b = (const float*)d_in[9];
    const float* toz_w  = (const float*)d_in[10];
    const float* toz_b  = (const float*)d_in[11];
    const float* vp_w1  = (const float*)d_in[12];
    const float* vp_b1  = (const float*)d_in[13];
    const float* vp_w2  = (const float*)d_in[14];
    const float* vp_b2  = (const float*)d_in[15];
    const float* br_w   = (const float*)d_in[16];
    const float* br_b   = (const float*)d_in[17];

    // workspace: agg[N_NODES*16] | cnt[N_NODES] | pooled[1024*32]  (~17.1 MB)
    float* agg    = (float*)d_ws;
    float* cnt    = agg + (size_t)N_NODES * 16;
    float* pooled = cnt + N_NODES;
    const int ws_elems = N_NODES * 16 + N_NODES + NUM_GRAPHS * 32;

    // output: recon | z | s | forces | V
    float* out    = (float*)d_out;
    float* recon  = out;
    float* z      = out + 32768;
    float* s      = out + 32768 + 8192;
    float* forces = out + 32768 + 8192 + 524288;
    float* V      = out + 32768 + 8192 + 524288 + 4096;

    zero_f32<<<2048, 256, 0, stream>>>((float*)d_ws, ws_elems);

    edge_mlp_wmma<<<EDGE_BLOCKS, 256, 0, stream>>>(
        x, pos, ei, enc_w1, enc_b1, enc_w2, enc_b2, agg, cnt);

    node_pool<<<N_NODES / 256, 256, 0, stream>>>(
        agg, cnt, batch, pool_w, pool_b, s, pooled);

    graph_head<<<NUM_GRAPHS / 256, 256, 0, stream>>>(
        pooled, toz_w, toz_b, vp_w1, vp_b1, vp_w2, vp_b2, br_w, br_b,
        recon, z, forces, V);
}
